// IncConcaveNN_35734127903350
// MI455X (gfx1250) — matrix operand order repaired and checked by reading, not compiled, not run on hardware
//
#include <hip/hip_runtime.h>
#include <math.h>

#define NB_   50
#define S1_   51
#define IND_  64
#define HID_  128
#define B_    256
#define A_K   136   // padded f16 row stride for the LDS A tile (128 + 8)
#define IC_   3     // outer-index chunk per block (51 = 17 * 3)
#define NCH_  17

typedef __attribute__((ext_vector_type(16))) _Float16 v16h;
typedef __attribute__((ext_vector_type(8)))  _Float16 v8h;
typedef __attribute__((ext_vector_type(8)))  float    v8f;

// ---------------- workspace layout (float offsets) ----------------
#define WS_STEPS   0                  // 64
#define WS_CC      64                 // 64
#define WS_XMAX    128                // 1 (padded)
#define WS_OFFSET  192                // 256
#define WS_SCALE   448                // 256
#define WS_BASE    704                // 256*128 = 32768
#define WS_CONTRIB 33472              // 256*64  = 16384
#define WS_WT      49856              // 128*128 f16 = 8192 floats (16B aligned)

static __device__ __forceinline__ v16h pack16(v8h lo, v8h hi) {
  return __builtin_shufflevector(lo, hi, 0,1,2,3,4,5,6,7,8,9,10,11,12,13,14,15);
}

// ---------------- setup: steps, cc (float64 like numpy), xmax ----------------
__global__ __launch_bounds__(256) void k_prep(const float* __restrict__ x,
                                              float* __restrict__ ws) {
  __shared__ float s_steps[S1_];
  __shared__ float s_max[256];
  const double PI = 3.14159265358979323846;
  int t = threadIdx.x;
  if (t < S1_) {
    double st = cos((double)t * PI / (double)NB_);
    s_steps[t] = (float)st;
    ws[WS_STEPS + t] = (float)st;
    // cc[t] = sum_i W[i] * lam[i][t] ; only even i contribute
    double s = 0.0;
    for (int i = 0; i <= NB_; i += 2) {
      double Wi = (i == 0) ? 1.0 : 2.0 / (1.0 - (double)(i * i));
      double m;
      if (t == 0 || t == NB_) m = 0.5;   // lam[:,0]=0.5 ; lam[:,NB]=0.5*cos(i*pi)=0.5 (i even)
      else m = cos((double)(i * t) * PI / (double)NB_);
      s += Wi * m;
    }
    ws[WS_CC + t] = (float)(s * 2.0 / (double)NB_);
  }
  __syncthreads();
  float xv = x[t];
  float m = -1e30f;
  for (int i = 0; i < S1_; ++i) m = fmaxf(m, xv * (s_steps[i] + 1.f) * 0.5f);
  s_max[t] = m;
  __syncthreads();
  if (t == 0) {
    float mm = -1e30f;
    for (int i = 0; i < 256; ++i) mm = fmaxf(mm, s_max[i]);
    ws[WS_XMAX] = mm + 10.f;
  }
}

// ------ per-batch: base = ib0 + h@iw0[1:], and the small noise MLP ------
__global__ __launch_bounds__(128) void k_base_noise(
    const float* __restrict__ h,   const float* __restrict__ iw0,
    const float* __restrict__ ib0, const float* __restrict__ nw0,
    const float* __restrict__ nb0, const float* __restrict__ nw1,
    const float* __restrict__ nb1, const float* __restrict__ nw2,
    const float* __restrict__ nb2, float* __restrict__ ws) {
  __shared__ float sh[IND_ - 1];
  __shared__ float so[HID_];
  int b = blockIdx.x, k = threadIdx.x;
  if (k < IND_ - 1) sh[k] = h[b * (IND_ - 1) + k];
  __syncthreads();
  float accb = ib0[k];
  float acc0 = nb0[k];
  for (int d = 0; d < IND_ - 1; ++d) {
    float hv = sh[d];
    accb = fmaf(hv, iw0[(d + 1) * HID_ + k], accb);
    acc0 = fmaf(hv, nw0[d * HID_ + k], acc0);
  }
  ws[WS_BASE + b * HID_ + k] = accb;
  so[k] = fmaxf(acc0, 0.f);
  __syncthreads();
  float acc1 = nb1[k];
  for (int d = 0; d < HID_; ++d) acc1 = fmaf(so[d], nw1[d * HID_ + k], acc1);
  __syncthreads();
  so[k] = fmaxf(acc1, 0.f);
  __syncthreads();
  if (k < 2) {
    float o = nb2[k];
    for (int d = 0; d < HID_; ++d) o = fmaf(so[d], nw2[d * 2 + k], o);
    if (k == 0) ws[WS_OFFSET + b] = o;
    else        ws[WS_SCALE + b] = expf(o);
  }
}

// ------ transpose iw1 -> N-major f16 so WMMA B fragments are contiguous ------
__global__ __launch_bounds__(256) void k_wt(const float* __restrict__ iw1,
                                            float* __restrict__ ws) {
  _Float16* wt = (_Float16*)(ws + WS_WT);
  int idx = blockIdx.x * blockDim.x + threadIdx.x;   // 0..16383
  int n = idx >> 7, k = idx & 127;
  wt[n * HID_ + k] = (_Float16)iw1[k * HID_ + n];    // wt[n][k] = W1[k][n]
}

// ---------------- main fused kernel: one block per (i-chunk, b) ----------------
__global__ __launch_bounds__(256) void k_main(
    const float* __restrict__ x,   const float* __restrict__ iw0,
    const float* __restrict__ ib1, const float* __restrict__ iw2,
    const float* __restrict__ ib2, float* __restrict__ ws) {
  __shared__ _Float16 sA[64 * A_K];       // 64 rows (51 real) x 128 cols, f16
  __shared__ float    sPartial[8 * 64];   // per-wave layer-2 row partials
  __shared__ float    sRed[64];
  __shared__ float    sBase[HID_];        // ib0 + h_b @ iw0[1:,:]
  __shared__ float    sW0[HID_];          // iw0 row 0 (the s-weight)
  __shared__ float    sSteps[S1_];

  const float*    cc      = ws + WS_CC;
  const _Float16* wt      = (const _Float16*)(ws + WS_WT);
  float*          contrib = ws + WS_CONTRIB;

  const int tid  = threadIdx.x;
  const int lane = tid & 31;
  const int w    = tid >> 5;      // wave id 0..7 -> owns column tile w
  const int i0   = blockIdx.x * IC_;
  const int b    = blockIdx.y;    // batch

  const int nl   = lane & 15;
  const int half = lane >> 4;
  const int ncol = w * 16 + nl;

  __builtin_prefetch(wt + ncol * HID_, 0, 1);   // global_prefetch_b8

  // ---- stage i-invariant data ----
  if (tid < HID_) {
    sW0[tid]   = iw0[tid];
    sBase[tid] = ws[WS_BASE + b * HID_ + tid];
  }
  if (tid < S1_) sSteps[tid] = ws[WS_STEPS + tid];

  const float xmax = ws[WS_XMAX];
  const float xb   = x[b];

  // ---- B fragments for this wave's column tile, global -> VGPRs, held all loop.
  // Contiguous k order used here (A matches): lane half h holds K = h*16 + 0..15 (pairs per VGPR), lane%16 = N.
  v16h bfrag[4];
#pragma unroll
  for (int kk = 0; kk < 4; ++kk) {
    const _Float16* p = wt + ncol * HID_ + kk * 32 + half * 16;
    bfrag[kk] = pack16(*(const v8h*)p, *(const v8h*)(p + 8));
  }

  const float biasn = ib1[ncol];
  const float w2n   = iw2[ncol];
  const float ib2v  = ib2[0];

  const int rowF = tid >> 2;                 // layer-0 fill: 4 threads per row
  const int kqF  = (tid & 3) * 32;

  __syncthreads();                            // sBase/sW0/sSteps visible

  for (int ii = 0; ii < IC_; ++ii) {
    const int i = i0 + ii;
    const float ti = xb * (sSteps[i] + 1.f) * 0.5f;

    // ---- layer 0 (rank-1 + precomputed base): fill a0 tile in LDS as f16 ----
    {
      float sv = (rowF < S1_) ? (ti + (xmax - ti) * (sSteps[rowF] + 1.f) * 0.5f) : 0.f;
#pragma unroll
      for (int g = 0; g < 4; ++g) {
        v8h pk;
#pragma unroll
        for (int e = 0; e < 8; ++e) {
          int k = kqF + g * 8 + e;
          float v = fmaxf(fmaf(sv, sW0[k], sBase[k]), 0.f);
          pk[e] = (_Float16)v;
        }
        *(v8h*)&sA[rowF * A_K + kqF + g * 8] = pk;
      }
    }
    __syncthreads();   // B1: a0 tile ready

    // ---- layer 1 GEMM: 4 row tiles x (K=128 as 4 x K32) WMMA, bias preloaded in C
    v8f acc[4];
#pragma unroll
    for (int rt = 0; rt < 4; ++rt)
#pragma unroll
      for (int e = 0; e < 8; ++e) acc[rt][e] = biasn;

#pragma unroll
    for (int rt = 0; rt < 4; ++rt) {
      const int r = rt * 16 + nl;
#pragma unroll
      for (int kk = 0; kk < 4; ++kk) {
        // A layout (matches B): half 0 -> K = 0..15; half 1 -> K = 16..31
        const _Float16* p = &sA[r * A_K + kk * 32 + half * 16];
        v16h afrag = pack16(*(const v8h*)p, *(const v8h*)(p + 8));
        acc[rt] = __builtin_amdgcn_wmma_f32_16x16x32_f16(
            false, afrag, false, bfrag[kk], (short)0, acc[rt], false, false);
      }
    }

    // ---- relu + layer-2 dot (w2), butterfly over the 16 lanes sharing a row ----
#pragma unroll
    for (int rt = 0; rt < 4; ++rt) {
#pragma unroll
      for (int v = 0; v < 8; ++v) {
        float tval = fmaxf(acc[rt][v], 0.f) * w2n;
        tval += __shfl_xor(tval, 1);
        tval += __shfl_xor(tval, 2);
        tval += __shfl_xor(tval, 4);
        tval += __shfl_xor(tval, 8);
        if (nl == 0) sPartial[w * 64 + rt * 16 + half * 8 + v] = tval;  // C layout: M = v + 8*half
      }
    }
    __syncthreads();   // B2: all waves done reading sA, partials published

    // ---- deterministic cross-wave sum, ELU(+1), cc[j] weight ----
    if (tid < 64) {
      float z = 0.f;
#pragma unroll
      for (int ww = 0; ww < 8; ++ww) z += sPartial[ww * 64 + tid];
      z += ib2v;
      float f = (z > 0.f) ? (z + 1.f) : expf(z);   // elu(z)+1
      sRed[tid] = (tid < S1_) ? f * cc[tid] : 0.f;
    }
    __syncthreads();   // B3: sRed ready
    if (tid == 0) {
      float s = 0.f;
      for (int j = 0; j < S1_; ++j) s += sRed[j];
      contrib[b * 64 + i] = s * (xmax - ti) * 0.5f * cc[i];
    }
  }
}

// ---------------- finalize: outer sum + scaling/offset ----------------
__global__ __launch_bounds__(256) void k_final(const float* __restrict__ x,
                                               const float* __restrict__ ws,
                                               float* __restrict__ out) {
  int b = threadIdx.x;
  const float* contrib = ws + WS_CONTRIB;
  float s = 0.f;
  for (int i = 0; i < S1_; ++i) s += contrib[b * 64 + i];
  out[b] = ws[WS_SCALE + b] * s * x[b] * 0.5f + ws[WS_OFFSET + b];
}

extern "C" void kernel_launch(void* const* d_in, const int* in_sizes, int n_in,
                              void* d_out, int out_size, void* d_ws, size_t ws_size,
                              hipStream_t stream) {
  const float* x   = (const float*)d_in[0];
  const float* h   = (const float*)d_in[1];
  const float* iw0 = (const float*)d_in[2];
  const float* ib0 = (const float*)d_in[3];
  const float* iw1 = (const float*)d_in[4];
  const float* ib1 = (const float*)d_in[5];
  const float* iw2 = (const float*)d_in[6];
  const float* ib2 = (const float*)d_in[7];
  const float* nw0 = (const float*)d_in[8];
  const float* nb0 = (const float*)d_in[9];
  const float* nw1 = (const float*)d_in[10];
  const float* nb1 = (const float*)d_in[11];
  const float* nw2 = (const float*)d_in[12];
  const float* nb2 = (const float*)d_in[13];
  float* ws  = (float*)d_ws;
  float* out = (float*)d_out;

  hipLaunchKernelGGL(k_prep, dim3(1), dim3(256), 0, stream, x, ws);
  hipLaunchKernelGGL(k_base_noise, dim3(256), dim3(128), 0, stream,
                     h, iw0, ib0, nw0, nb0, nw1, nb1, nw2, nb2, ws);
  hipLaunchKernelGGL(k_wt, dim3(64), dim3(256), 0, stream, iw1, ws);
  hipLaunchKernelGGL(k_main, dim3(NCH_, B_), dim3(256), 0, stream,
                     x, iw0, ib1, iw2, ib2, ws);
  hipLaunchKernelGGL(k_final, dim3(1), dim3(256), 0, stream, x, ws, out);
}
